// TTTInner_61778809586253
// MI455X (gfx1250) — matrix operand order repaired and checked by reading, compiled, not run on hardware
//
#include <hip/hip_runtime.h>

typedef __bf16 bf16_t;
typedef bf16_t v16bf __attribute__((ext_vector_type(16)));
typedef float  v8f   __attribute__((ext_vector_type(8)));
typedef unsigned short ushort_t;
typedef ushort_t us8 __attribute__((ext_vector_type(8)));
typedef ushort_t us4 __attribute__((ext_vector_type(4)));

#define FDIM 256
#define TOK  2048          // tokens per minibatch step (MB*B = 64*32)
#define NMB  64            // number of sequential steps
#define INV_N (2.0f / 524288.0f)   // 2/(MB*B*F)

__device__ __forceinline__ ushort_t f2bf(float f) {
  unsigned u = __builtin_bit_cast(unsigned, f);
  unsigned r = 0x7FFFu + ((u >> 16) & 1u);
  return (ushort_t)((u + r) >> 16);
}
__device__ __forceinline__ float bf2f(ushort_t h) {
  unsigned u = ((unsigned)h) << 16;
  return __builtin_bit_cast(float, u);
}

union Frag2 { v16bf v; us8 h[2]; ushort_t u[16]; };

// A fragment (16x32 bf16) from a row-major (rows x K) bf16 matrix `p`
// (pre-offset to the tile's first row). ISA layout: lane half g holds
// K = k0+8g+{0..7} (v0..3) and k0+16+8g+{0..7} (v4..7) -> two 16B runs.
__device__ __forceinline__ v16bf loadA_rm(const ushort_t* p, int ld, int k0, int lane) {
  int r = lane & 15;
  int g = (lane >> 4) & 1;
  const ushort_t* row = p + (size_t)r * ld + k0 + 8 * g;
  Frag2 f;
  f.h[0] = *(const us8*)(row);
  f.h[1] = *(const us8*)(row + 16);
  return f.v;
}

// B fragment (32x16 bf16) where B[k,n] = p[n*ld + k] (i.e. p holds B^T
// row-major, K contiguous). ISA layout: lane half g holds K = k0+8g+{0..7}
// (v0..3) and k0+16+8g+{0..7} (v4..7) -> two 16B runs.
__device__ __forceinline__ v16bf loadBT_rm(const ushort_t* p, int ld, int k0, int n0, int lane) {
  int c = n0 + (lane & 15);
  int g = (lane >> 4) & 1;
  const ushort_t* row = p + (size_t)c * ld + k0 + 8 * g;
  Frag2 f;
  f.h[0] = *(const us8*)(row);
  f.h[1] = *(const us8*)(row + 16);
  return f.v;
}

#define WMMA_BF16(A, B, C) \
  __builtin_amdgcn_wmma_f32_16x16x32_bf16(false, (A), false, (B), (short)0, (C), false, false)

// ---------------------------------------------------------------------------
// One-time init: TRANSPOSED bf16 copies of theta_k/q/v (K-contiguous for B
// fragments); row-major fp32 W + bf16 shadow; b; zero accumulators.
__global__ void __launch_bounds__(256) kernel_init(
    const float* __restrict__ thKf, const float* __restrict__ thQf,
    const float* __restrict__ thVf, const float* __restrict__ W0,
    const float* __restrict__ b0,
    float* __restrict__ W, ushort_t* __restrict__ thKT, ushort_t* __restrict__ thQT,
    ushort_t* __restrict__ thVT, ushort_t* __restrict__ Wbf,
    float* __restrict__ bvec, float* __restrict__ gb, float* __restrict__ xsum) {
  int e = blockIdx.x * 256 + threadIdx.x;
  int k = e >> 8, n = e & 255;
  int t = n * FDIM + k;
  thKT[t] = f2bf(thKf[e]);
  thQT[t] = f2bf(thQf[e]);
  thVT[t] = f2bf(thVf[e]);
  float w = W0[e];
  W[e] = w;
  Wbf[e] = f2bf(w);
  if (e < FDIM) { bvec[e] = b0[e]; gb[e] = 0.f; xsum[e] = 0.f; }
}

// ---------------------------------------------------------------------------
// Per step: train/label/test views + diff, plus sum-of-x accumulation.
// Grid: 128 WGs (16-token tiles) x 128 threads (4 waves x 4 N-tiles each).
// Outputs trainT/diffT (256 x TOK bf16) and testT (256 x TOK fp32).
__global__ void __launch_bounds__(128) kernel_views(
    const float* __restrict__ src,
    const ushort_t* __restrict__ thKT, const ushort_t* __restrict__ thQT,
    const ushort_t* __restrict__ thVT, const ushort_t* __restrict__ Wbf,
    const float* __restrict__ bvec,
    float* __restrict__ testT, ushort_t* __restrict__ trainT,
    ushort_t* __restrict__ diffT, float* __restrict__ xsum, int step) {
  __shared__ __align__(16) ushort_t xls[16 * FDIM];
  __shared__ __align__(16) ushort_t trls[16 * FDIM];
  int tid = threadIdx.x;
  int m0 = blockIdx.x * 16;
  const float* xsrc = src + ((size_t)step * TOK + m0) * FDIM;

  // Stage x tile (fp32 -> bf16 LDS) with float4 loads; column sums for xmean.
  float sx = 0.f, sy = 0.f, sz = 0.f, sw = 0.f;
#pragma unroll
  for (int it = 0; it < 8; ++it) {
    int idx4 = (it * 128 + tid) * 4;
    float4 v = *(const float4*)(xsrc + idx4);
    sx += v.x; sy += v.y; sz += v.z; sw += v.w;   // col = (4*tid)&255, fixed
    us4 pk;
    pk[0] = f2bf(v.x); pk[1] = f2bf(v.y); pk[2] = f2bf(v.z); pk[3] = f2bf(v.w);
    *(us4*)(xls + idx4) = pk;
  }
  int c0 = (4 * tid) & 255;
  atomicAdd(&xsum[c0 + 0], sx);
  atomicAdd(&xsum[c0 + 1], sy);
  atomicAdd(&xsum[c0 + 2], sz);
  atomicAdd(&xsum[c0 + 3], sw);
  __syncthreads();

  int wave = tid >> 5, lane = tid & 31;
  int c = lane & 15, g = (lane >> 4) & 1;
  v8f tgt[4];
#pragma unroll
  for (int nt = 0; nt < 4; ++nt) {
    int n0 = (wave * 4 + nt) * 16;
    v8f aK = {}; v8f aV = {}; v8f aQ = {};
    for (int k0 = 0; k0 < FDIM; k0 += 32) {
      v16bf a = loadA_rm(xls, FDIM, k0, lane);
      aK = WMMA_BF16(a, loadBT_rm(thKT, FDIM, k0, n0, lane), aK);
      aV = WMMA_BF16(a, loadBT_rm(thVT, FDIM, k0, n0, lane), aV);
      aQ = WMMA_BF16(a, loadBT_rm(thQT, FDIM, k0, n0, lane), aQ);
    }
    tgt[nt] = aV - aK;                       // target = label - train
    int n = n0 + c;
    // C tile: lane holds column n, rows m0+8g+{0..7} -> contiguous stores.
    us8 tb;
#pragma unroll
    for (int v = 0; v < 8; ++v) tb[v] = f2bf(aK[v]);
    *(us8*)(trainT + (size_t)n * TOK + m0 + 8 * g) = tb;
#pragma unroll
    for (int v = 0; v < 8; ++v) trls[(v + 8 * g) * FDIM + n] = tb[v];
    float4 q0 = { aQ[0], aQ[1], aQ[2], aQ[3] };
    float4 q1 = { aQ[4], aQ[5], aQ[6], aQ[7] };
    float* tout = testT + (size_t)n * TOK + m0 + 8 * g;
    *(float4*)(tout)     = q0;
    *(float4*)(tout + 4) = q1;
  }
  __syncthreads();

  // diff = train @ W^T + b - target
#pragma unroll
  for (int nt = 0; nt < 4; ++nt) {
    int n0 = (wave * 4 + nt) * 16;
    int n = n0 + c;
    float bn = bvec[n];
    v8f aD = {};
    for (int k0 = 0; k0 < FDIM; k0 += 32) {
      v16bf a = loadA_rm(trls, FDIM, k0, lane);
      aD = WMMA_BF16(a, loadBT_rm(Wbf, FDIM, k0, n0, lane), aD);
    }
    us8 db;
#pragma unroll
    for (int v = 0; v < 8; ++v) db[v] = f2bf(aD[v] + bn - tgt[nt][v]);
    *(us8*)(diffT + (size_t)n * TOK + m0 + 8 * g) = db;
  }
}

// ---------------------------------------------------------------------------
// Per step: gW = diff^T @ train (256x256, K=2048), gb folded into j0==0 tiles.
// A = diffT rows (K-contiguous), B = trainT (K-contiguous) -> all b128 loads.
// Grid: 256 WGs x 32 threads (one wave = one 16x16 tile).
__global__ void __launch_bounds__(32) kernel_gw(
    const ushort_t* __restrict__ diffT, const ushort_t* __restrict__ trainT,
    float* __restrict__ gW, float* __restrict__ gb) {
  int lane = threadIdx.x;
  int i0 = (blockIdx.x & 15) * 16;
  int j0 = (blockIdx.x >> 4) * 16;
  int c = lane & 15, g = (lane >> 4) & 1;
  v8f acc = {};
  float gbacc = 0.f;
  const ushort_t* arow = diffT + (size_t)i0 * TOK;
  for (int k0 = 0; k0 < TOK; k0 += 32) {
    Frag2 fa;
    {
      const ushort_t* row = arow + (size_t)(lane & 15) * TOK + k0 + 8 * g;
      fa.h[0] = *(const us8*)(row);
      fa.h[1] = *(const us8*)(row + 16);
    }
    v16bf bT = loadBT_rm(trainT, TOK, k0, j0, lane);
    acc = WMMA_BF16(fa.v, bT, acc);
    if (j0 == 0) {  // gb[i] = sum_k diff[k,i]; lane halves cover disjoint K
#pragma unroll
      for (int h = 0; h < 16; ++h) gbacc += bf2f(fa.u[h]);
    }
  }
#pragma unroll
  for (int v = 0; v < 8; ++v)
    gW[(size_t)(i0 + v + 8 * g) * FDIM + j0 + c] = acc[v];
  if (j0 == 0) atomicAdd(&gb[i0 + c], gbacc);
}

// ---------------------------------------------------------------------------
// Per step: lr GEMV against W_lr + weight/bias update.
// Grid: 256 WGs (one weight row) x 256 threads (one column each).
__global__ void __launch_bounds__(256) kernel_update(
    const float* __restrict__ xsum, const float* __restrict__ W_lr,
    const float* __restrict__ gWraw, const float* __restrict__ gb,
    float* __restrict__ W, ushort_t* __restrict__ Wbf, float* __restrict__ bvec) {
  __shared__ float xm[FDIM];
  __shared__ float red[FDIM];
  int r = blockIdx.x, c = threadIdx.x;
  xm[c] = xsum[c] * (1.0f / (float)TOK);
  __syncthreads();
  size_t e = (size_t)r * FDIM + c;
  float lr = 0.f;
#pragma unroll 4
  for (int i = 0; i < FDIM; ++i)
    lr += xm[i] * W_lr[(size_t)i * (FDIM * FDIM) + e];   // coalesced over c
  float wn = W[e] - lr * (INV_N * gWraw[e]);
  W[e] = wn;
  Wbf[e] = f2bf(wn);
  red[c] = lr;
  __syncthreads();
  for (int s = 128; s > 0; s >>= 1) {
    if (c < s) red[c] += red[c + s];
    __syncthreads();
  }
  if (c == 0) {
    float lrb = red[0] * (1.0f / (float)FDIM);
    bvec[r] = bvec[r] - lrb * (INV_N * gb[r]);
  }
}

// ---------------------------------------------------------------------------
// Per step: z = test @ Wn^T + bn + test -> out; zero accumulators for next step.
// Grid: 128 WGs x 128 threads.
__global__ void __launch_bounds__(128) kernel_z(
    const float* __restrict__ testT, const ushort_t* __restrict__ Wbf,
    const float* __restrict__ bvec, float* __restrict__ out,
    float* __restrict__ gb, float* __restrict__ xsum, int step) {
  __shared__ __align__(16) ushort_t tls[16 * FDIM];
  int tid = threadIdx.x;
  int m0 = blockIdx.x * 16;
  if (blockIdx.x == 0) {   // safe: this kernel never reads gb/xsum
    gb[tid] = 0.f; gb[tid + 128] = 0.f;
    xsum[tid] = 0.f; xsum[tid + 128] = 0.f;
  }
  // Stage test tile: testT is (256 x TOK) fp32; tile = rows f, cols m0..m0+15.
#pragma unroll
  for (int it = 0; it < 8; ++it) {
    int idx = it * 128 + tid;          // 0..1023
    int f = idx >> 2;                  // 0..255
    int t = (idx & 3) * 4;             // 0,4,8,12
    float4 v = *(const float4*)(testT + (size_t)f * TOK + m0 + t);
    tls[(t + 0) * FDIM + f] = f2bf(v.x);
    tls[(t + 1) * FDIM + f] = f2bf(v.y);
    tls[(t + 2) * FDIM + f] = f2bf(v.z);
    tls[(t + 3) * FDIM + f] = f2bf(v.w);
  }
  __syncthreads();
  int wave = tid >> 5, lane = tid & 31;
  int c = lane & 15, g = (lane >> 4) & 1;
  float* zo = out + ((size_t)step * TOK + m0) * FDIM;
#pragma unroll
  for (int nt = 0; nt < 4; ++nt) {
    int n0 = (wave * 4 + nt) * 16;
    int n = n0 + c;
    float bn = bvec[n];
    v8f acc = {};
    for (int k0 = 0; k0 < FDIM; k0 += 32) {
      v16bf a = loadA_rm(tls, FDIM, k0, lane);
      acc = WMMA_BF16(a, loadBT_rm(Wbf, FDIM, k0, n0, lane), acc);
    }
    const float* trow = testT + (size_t)n * TOK + m0 + 8 * g;
    float4 r0 = *(const float4*)(trow);
    float4 r1 = *(const float4*)(trow + 4);
    float res[8] = { r0.x, r0.y, r0.z, r0.w, r1.x, r1.y, r1.z, r1.w };
#pragma unroll
    for (int v = 0; v < 8; ++v)
      zo[(size_t)(v + 8 * g) * FDIM + n] = acc[v] + bn + res[v];
  }
}

// ---------------------------------------------------------------------------
extern "C" void kernel_launch(void* const* d_in, const int* in_sizes, int n_in,
                              void* d_out, int out_size, void* d_ws, size_t ws_size,
                              hipStream_t stream) {
  (void)in_sizes; (void)n_in; (void)out_size; (void)ws_size;
  const float* src    = (const float*)d_in[0];
  const float* thKf   = (const float*)d_in[1];
  const float* thQf   = (const float*)d_in[2];
  const float* thVf   = (const float*)d_in[3];
  const float* W0     = (const float*)d_in[4];
  const float* b0     = (const float*)d_in[5];
  const float* W_lr   = (const float*)d_in[6];
  float* out = (float*)d_out;

  // Workspace layout (fp32 region then bf16 region), ~5.25 MB total.
  float* W     = (float*)d_ws;            // 65536
  float* bvec  = W + 65536;               // 256
  float* gb    = bvec + 256;              // 256
  float* xsum  = gb + 256;                // 256
  float* gW    = xsum + 256;              // 65536
  float* testT = gW + 65536;              // 524288 (256 x TOK fp32)
  ushort_t* thKT   = (ushort_t*)(testT + 524288);
  ushort_t* thQT   = thKT + 65536;
  ushort_t* thVT   = thQT + 65536;
  ushort_t* Wbf    = thVT + 65536;        // row-major (already K-contiguous for W^T)
  ushort_t* trainT = Wbf + 65536;         // 256 x TOK bf16
  ushort_t* diffT  = trainT + 524288;     // 256 x TOK bf16

  kernel_init<<<256, 256, 0, stream>>>(thKf, thQf, thVf, W0, b0,
                                       W, thKT, thQT, thVT, Wbf, bvec, gb, xsum);
  for (int s = 0; s < NMB; ++s) {
    kernel_views<<<128, 128, 0, stream>>>(src, thKT, thQT, thVT, Wbf, bvec,
                                          testT, trainT, diffT, xsum, s);
    kernel_gw<<<256, 32, 0, stream>>>(diffT, trainT, gW, gb);
    kernel_update<<<256, 256, 0, stream>>>(xsum, W_lr, gW, gb, W, Wbf, bvec);
    kernel_z<<<128, 128, 0, stream>>>(testT, Wbf, bvec, out, gb, xsum, s);
  }
}
